// EncoderBlock_1717986918591
// MI455X (gfx1250) — compile-verified, matrix operand order checked by reading, not run on hardware
//
#include <hip/hip_runtime.h>
#include <hip/hip_bf16.h>

// ---------------------------------------------------------------------------
// Types for CDNA5 WMMA (gfx1250, wave32)
// ---------------------------------------------------------------------------
typedef __attribute__((ext_vector_type(16))) __bf16 v16bf;
typedef __attribute__((ext_vector_type(8)))  float  v8f;

union FragU {
    uint4 u[2];
    v16bf v;
};

__device__ __forceinline__ unsigned short f2bf(float f) {
    // round-to-nearest-even fp32 -> bf16
    unsigned int u = __float_as_uint(f);
    u += 0x7FFFu + ((u >> 16) & 1u);
    return (unsigned short)(u >> 16);
}

// Load a 16x32 bf16 A-fragment (or B-fragment from a pre-transposed Bt matrix)
// following the CDNA5 16-bit A-matrix VGPR layout:
//   lane l: row m = l&15, half = l>>4
//   VGPR0..3 : K = half*8 .. half*8+7       (16 bytes)
//   VGPR4..7 : K = 16+half*8 .. 16+half*8+7 (16 bytes)
__device__ __forceinline__ v16bf load_frag(const unsigned short* base, int ld, int lane) {
    const int m    = lane & 15;
    const int half = (lane >> 4) & 1;
    const unsigned short* p = base + (size_t)m * ld + half * 8;
    FragU f;
    f.u[0] = *(const uint4*)(p);
    f.u[1] = *(const uint4*)(p + 16);
    return f.v;
}

__device__ __forceinline__ v8f wmma_bf16(v16bf a, v16bf b, v8f c) {
    // (neg_a, A, neg_b, B, c_mod, C, reuse_a, reuse_b)
    return __builtin_amdgcn_wmma_f32_16x16x32_bf16(false, a, false, b, (short)0, c, false, false);
}

// 16-lane max reduction entirely in VALU via DPP16 (no LDS round trips).
// Steps: xor1 (quad_perm 1,0,3,2), xor2 (quad_perm 2,3,0,1),
//        row_half_mirror (exchange quads in octet), row_mirror (exchange octets).
template <int CTRL>
__device__ __forceinline__ float dpp_fmax_step(float x) {
    int y = __builtin_amdgcn_update_dpp(0, __float_as_int(x), CTRL, 0xF, 0xF, true);
    return fmaxf(x, __int_as_float(y));
}
__device__ __forceinline__ float rowmax16(float x) {
    x = dpp_fmax_step<0xB1>(x);   // quad_perm(1,0,3,2)
    x = dpp_fmax_step<0x4E>(x);   // quad_perm(2,3,0,1)
    x = dpp_fmax_step<0x141>(x);  // row_half_mirror
    x = dpp_fmax_step<0x140>(x);  // row_mirror
    return x;
}

// ---------------------------------------------------------------------------
// Tiled weight convert + transpose: Wt[n*K + k] = bf16(W[k*N + n])
// ---------------------------------------------------------------------------
__global__ __launch_bounds__(256) void transpose_f32_to_bf16(
    const float* __restrict__ W, unsigned short* __restrict__ Wt, int K, int N) {
    __shared__ float t[32][33];
    const int n0 = blockIdx.x * 32;
    const int k0 = blockIdx.y * 32;
    const int tx = threadIdx.x & 31;
    const int ty = threadIdx.x >> 5;  // 0..7
#pragma unroll
    for (int i = 0; i < 32; i += 8)
        t[ty + i][tx] = W[(size_t)(k0 + ty + i) * N + n0 + tx];
    __syncthreads();
#pragma unroll
    for (int i = 0; i < 32; i += 8)
        Wt[(size_t)(n0 + ty + i) * K + k0 + tx] = f2bf(t[tx][ty + i]);
}

// ---------------------------------------------------------------------------
// LayerNorm over 1024 columns, one row per block, bf16 output
// ---------------------------------------------------------------------------
__global__ __launch_bounds__(256) void layernorm_bf16(
    const float* __restrict__ x, const float* __restrict__ g,
    const float* __restrict__ be, unsigned short* __restrict__ out) {
    const int row  = blockIdx.x;
    const int lane = threadIdx.x & 31;
    const int wave = threadIdx.x >> 5;
    const float* xr = x + (size_t)row * 1024;

    float v[4];
    float s = 0.f, ss = 0.f;
#pragma unroll
    for (int i = 0; i < 4; ++i) {
        int idx = threadIdx.x + i * 256;
        v[i] = xr[idx];
        s  += v[i];
        ss += v[i] * v[i];
    }
#pragma unroll
    for (int m = 1; m < 32; m <<= 1) {
        s  += __shfl_xor(s,  m);
        ss += __shfl_xor(ss, m);
    }
    __shared__ float sh_s[8], sh_ss[8];
    if (lane == 0) { sh_s[wave] = s; sh_ss[wave] = ss; }
    __syncthreads();
    float ts = 0.f, tss = 0.f;
#pragma unroll
    for (int w = 0; w < 8; ++w) { ts += sh_s[w]; tss += sh_ss[w]; }
    const float mu   = ts * (1.0f / 1024.0f);
    const float var  = tss * (1.0f / 1024.0f) - mu * mu;
    const float rstd = rsqrtf(var + 1e-5f);
#pragma unroll
    for (int i = 0; i < 4; ++i) {
        int idx = threadIdx.x + i * 256;
        out[(size_t)row * 1024 + idx] = f2bf((v[i] - mu) * rstd * g[idx] + be[idx]);
    }
}

// ---------------------------------------------------------------------------
// WMMA GEMM: C = out_scale * (A(MxK bf16 rm) @ Bt^T) + bias  (Bt is NxK rm)
// Block: 256 threads = 8 waves (4 row-waves x 2 col-waves),
// block tile 128x128, wave tile 32x64 (2x4 WMMA tiles).
// Fragments are double-buffered so next-iteration loads overlap the WMMAs.
// ---------------------------------------------------------------------------
constexpr int OUT_BF16      = 0;  // bf16 row-major
constexpr int OUT_BF16_T    = 1;  // bf16 transposed (ldc = M)
constexpr int OUT_F32_RES   = 2;  // f32 = resid + acc + bias
constexpr int OUT_BF16_RELU = 3;  // bf16 relu(acc + bias)

template <int MODE>
__global__ __launch_bounds__(256) void gemm_wmma(
    const unsigned short* __restrict__ A, int lda,
    const unsigned short* __restrict__ Bt, int ldb, int K,
    const float* __restrict__ bias, const float* __restrict__ resid,
    void* __restrict__ Cout, int ldc, float out_scale) {
    const int lane = threadIdx.x & 31;
    const int wave = threadIdx.x >> 5;
    const int wm = wave & 3;   // 0..3
    const int wn = wave >> 2;  // 0..1
    const int m0 = blockIdx.x * 128 + wm * 32;
    const int n0 = blockIdx.y * 128 + wn * 64;

    const unsigned short* A0 = A + (size_t)m0 * lda;
    const unsigned short* A1 = A + (size_t)(m0 + 16) * lda;
    const unsigned short* B0 = Bt + (size_t)n0 * ldb;
    const unsigned short* B1 = Bt + (size_t)(n0 + 16) * ldb;
    const unsigned short* B2 = Bt + (size_t)(n0 + 32) * ldb;
    const unsigned short* B3 = Bt + (size_t)(n0 + 48) * ldb;

    v8f acc[2][4] = {};
    v16bf a0 = load_frag(A0, lda, lane);
    v16bf a1 = load_frag(A1, lda, lane);
    v16bf b0 = load_frag(B0, ldb, lane);
    v16bf b1 = load_frag(B1, ldb, lane);
    v16bf b2 = load_frag(B2, ldb, lane);
    v16bf b3 = load_frag(B3, ldb, lane);

    for (int k = 0; k < K; k += 32) {
        const int kn = (k + 32 < K) ? (k + 32) : k;  // last iter: benign reload
        v16bf a0n = load_frag(A0 + kn, lda, lane);
        v16bf a1n = load_frag(A1 + kn, lda, lane);
        v16bf b0n = load_frag(B0 + kn, ldb, lane);
        v16bf b1n = load_frag(B1 + kn, ldb, lane);
        v16bf b2n = load_frag(B2 + kn, ldb, lane);
        v16bf b3n = load_frag(B3 + kn, ldb, lane);
        acc[0][0] = wmma_bf16(a0, b0, acc[0][0]);
        acc[0][1] = wmma_bf16(a0, b1, acc[0][1]);
        acc[0][2] = wmma_bf16(a0, b2, acc[0][2]);
        acc[0][3] = wmma_bf16(a0, b3, acc[0][3]);
        acc[1][0] = wmma_bf16(a1, b0, acc[1][0]);
        acc[1][1] = wmma_bf16(a1, b1, acc[1][1]);
        acc[1][2] = wmma_bf16(a1, b2, acc[1][2]);
        acc[1][3] = wmma_bf16(a1, b3, acc[1][3]);
        a0 = a0n; a1 = a1n; b0 = b0n; b1 = b1n; b2 = b2n; b3 = b3n;
    }

    // Epilogue. C/D layout: VGPR r, lanes 0-15 -> M=r, lanes 16-31 -> M=r+8; N = lane&15.
    const int cn    = lane & 15;
    const int rbase = (lane < 16) ? 0 : 8;
#pragma unroll
    for (int i = 0; i < 2; ++i)
#pragma unroll
        for (int j = 0; j < 4; ++j)
#pragma unroll
            for (int r = 0; r < 8; ++r) {
                const int row = m0 + i * 16 + rbase + r;
                const int col = n0 + j * 16 + cn;
                float v = acc[i][j][r] * out_scale;
                if (bias) v += bias[col];
                if constexpr (MODE == OUT_BF16) {
                    ((unsigned short*)Cout)[(size_t)row * ldc + col] = f2bf(v);
                } else if constexpr (MODE == OUT_BF16_T) {
                    ((unsigned short*)Cout)[(size_t)col * ldc + row] = f2bf(v);
                } else if constexpr (MODE == OUT_BF16_RELU) {
                    ((unsigned short*)Cout)[(size_t)row * ldc + col] = f2bf(fmaxf(v, 0.f));
                } else {
                    const float rv = resid[(size_t)row * ldc + col];
                    ((float*)Cout)[(size_t)row * ldc + col] = rv + v;
                }
            }
}

// ---------------------------------------------------------------------------
// Flash-attention: grid = (S/128, N*H), block = 128 (4 waves).
// Each wave: 32 q-rows (two 16-row tiles sharing K/V fragments),
// online softmax in base-2 over 32-key steps.
// Q is pre-scaled by log2(e)/sqrt(D) so p = exp2(s - m) directly.
// Per-half running max (exact: rows 0-7 live in lanes 0-15, rows 8-15 in
// lanes 16-31) reduced with 4 DPP steps -- no LDS shuffles.
// Row sums computed on the matrix unit via WMMA against an all-ones B frag.
// ---------------------------------------------------------------------------
__global__ __launch_bounds__(128) void attention_wmma(
    const unsigned short* __restrict__ Qm, const unsigned short* __restrict__ Km,
    const unsigned short* __restrict__ Vt, unsigned short* __restrict__ Out) {
    const int lane = threadIdx.x & 31;
    const int wave = threadIdx.x >> 5;
    const int bh = blockIdx.y;
    const int b  = bh >> 4;
    const int h  = bh & 15;
    const int q0 = blockIdx.x * 128 + wave * 32;
    const size_t rowQ = (size_t)b * 2048 + q0;

    const unsigned short* Qb = Qm + rowQ * 1024 + h * 64;
    v16bf qf[2][2];
#pragma unroll
    for (int t = 0; t < 2; ++t) {
        qf[t][0] = load_frag(Qb + (size_t)t * 16 * 1024, 1024, lane);
        qf[t][1] = load_frag(Qb + (size_t)t * 16 * 1024 + 32, 1024, lane);
    }

    // all-ones bf16 B fragment for row-sum WMMA
    FragU onesu;
#pragma unroll
    for (int i = 0; i < 2; ++i)
        onesu.u[i] = make_uint4(0x3F803F80u, 0x3F803F80u, 0x3F803F80u, 0x3F803F80u);
    const v16bf ones = onesu.v;

    float mi[2] = {-1e30f, -1e30f};  // running per-half max, per row-tile
    v8f o[2][4] = {};
    v8f lacc[2] = {};

    __shared__ __align__(16) unsigned short Pl[4][32][32];
    unsigned short (*P)[32] = Pl[wave];

    const unsigned short* Kbase = Km + (size_t)b * 2048 * 1024 + h * 64;
    const unsigned short* Vbase = Vt + (size_t)h * 64 * 4096 + (size_t)b * 2048;

    for (int kb = 0; kb < 2048; kb += 32) {
        // K fragments for 32 keys (shared by both q row tiles)
        const unsigned short* Kb0 = Kbase + (size_t)kb * 1024;
        const v16bf k0a = load_frag(Kb0, 1024, lane);
        const v16bf k0b = load_frag(Kb0 + 32, 1024, lane);
        const v16bf k1a = load_frag(Kb0 + (size_t)16 * 1024, 1024, lane);
        const v16bf k1b = load_frag(Kb0 + (size_t)16 * 1024 + 32, 1024, lane);

        v8f s[2][2];
#pragma unroll
        for (int t = 0; t < 2; ++t) {
            v8f z = {};
            s[t][0] = wmma_bf16(qf[t][0], k0a, z);
            s[t][0] = wmma_bf16(qf[t][1], k0b, s[t][0]);
            s[t][1] = wmma_bf16(qf[t][0], k1a, z);
            s[t][1] = wmma_bf16(qf[t][1], k1b, s[t][1]);
        }

#pragma unroll
        for (int t = 0; t < 2; ++t) {
            // per-half max (covers this half's 8 rows x all 32 keys), VALU-only
            float mx = fmaxf(s[t][0][0], s[t][1][0]);
#pragma unroll
            for (int r = 1; r < 8; ++r) mx = fmaxf(mx, fmaxf(s[t][0][r], s[t][1][r]));
            mx = rowmax16(mx);
            const float mnew  = fmaxf(mi[t], mx);
            const float alpha = exp2f(mi[t] - mnew);
            mi[t] = mnew;
#pragma unroll
            for (int r = 0; r < 8; ++r) {
                const float p0 = exp2f(s[t][0][r] - mnew);
                const float p1 = exp2f(s[t][1][r] - mnew);
                o[t][0][r] *= alpha; o[t][1][r] *= alpha;
                o[t][2][r] *= alpha; o[t][3][r] *= alpha;
                lacc[t][r] *= alpha;
                const int rr = t * 16 + ((lane < 16) ? r : (r + 8));
                P[rr][lane & 15]        = f2bf(p0);
                P[rr][(lane & 15) + 16] = f2bf(p1);
            }
        }
        asm volatile("s_wait_dscnt 0" ::: "memory");
        const v16bf pf0 = load_frag(&P[0][0], 32, lane);
        const v16bf pf1 = load_frag(&P[16][0], 32, lane);
        lacc[0] = wmma_bf16(pf0, ones, lacc[0]);
        lacc[1] = wmma_bf16(pf1, ones, lacc[1]);
#pragma unroll
        for (int c = 0; c < 4; ++c) {
            const v16bf vf = load_frag(Vbase + (size_t)c * 16 * 4096 + kb, 4096, lane);
            o[0][c] = wmma_bf16(pf0, vf, o[0][c]);
            o[1][c] = wmma_bf16(pf1, vf, o[1][c]);
        }
    }

    // finalize: divide by row sums, store bf16 in [4096,1024] (cols h*64..)
#pragma unroll
    for (int t = 0; t < 2; ++t)
#pragma unroll
        for (int c = 0; c < 4; ++c)
#pragma unroll
            for (int r = 0; r < 8; ++r) {
                const int rr  = t * 16 + ((lane < 16) ? r : (r + 8));
                const float inv = 1.0f / lacc[t][r];
                Out[(rowQ + rr) * 1024 + h * 64 + c * 16 + (lane & 15)] =
                    f2bf(o[t][c][r] * inv);
            }
}

// ---------------------------------------------------------------------------
// Host launcher
// ---------------------------------------------------------------------------
extern "C" void kernel_launch(void* const* d_in, const int* in_sizes, int n_in,
                              void* d_out, int out_size, void* d_ws, size_t ws_size,
                              hipStream_t stream) {
    constexpr int M   = 2 * 2048;  // 4096 tokens
    constexpr int DM  = 1024;
    constexpr int DFF = 4096;

    const float* x    = (const float*)d_in[0];
    const float* w_q  = (const float*)d_in[2];
    const float* w_k  = (const float*)d_in[3];
    const float* w_v  = (const float*)d_in[4];
    const float* w_0  = (const float*)d_in[5];
    const float* b_0  = (const float*)d_in[6];
    const float* w_1  = (const float*)d_in[7];
    const float* b_1  = (const float*)d_in[8];
    const float* w_2  = (const float*)d_in[9];
    const float* b_2  = (const float*)d_in[10];
    const float* g_1  = (const float*)d_in[11];
    const float* be_1 = (const float*)d_in[12];
    const float* g_2  = (const float*)d_in[13];
    const float* be_2 = (const float*)d_in[14];
    float* out = (float*)d_out;

    // workspace carve-up
    size_t off = 0;
    auto alloc = [&](size_t bytes) -> void* {
        void* p = (char*)d_ws + off;
        off += (bytes + 255) & ~(size_t)255;
        return p;
    };
    unsigned short* wq_t = (unsigned short*)alloc((size_t)DM * DM * 2);
    unsigned short* wk_t = (unsigned short*)alloc((size_t)DM * DM * 2);
    unsigned short* wv_t = (unsigned short*)alloc((size_t)DM * DM * 2);
    unsigned short* w0_t = (unsigned short*)alloc((size_t)DM * DM * 2);
    unsigned short* w1_t = (unsigned short*)alloc((size_t)DM * DFF * 2);  // [DFF][DM]
    unsigned short* w2_t = (unsigned short*)alloc((size_t)DFF * DM * 2);  // [DM][DFF]
    unsigned short* xn1  = (unsigned short*)alloc((size_t)M * DM * 2);
    unsigned short* Qb   = (unsigned short*)alloc((size_t)M * DM * 2);
    unsigned short* Kb   = (unsigned short*)alloc((size_t)M * DM * 2);
    unsigned short* Vt   = (unsigned short*)alloc((size_t)DM * M * 2);   // [DM][M]
    unsigned short* attn = (unsigned short*)alloc((size_t)M * DM * 2);
    float*          x1   = (float*)alloc((size_t)M * DM * 4);
    unsigned short* xn2  = (unsigned short*)alloc((size_t)M * DM * 2);
    unsigned short* hmid = (unsigned short*)alloc((size_t)M * DFF * 2);

    // 1) weights -> bf16 transposed (tiled, coalesced)
    transpose_f32_to_bf16<<<dim3(DM / 32, DM / 32), 256, 0, stream>>>(w_q, wq_t, DM, DM);
    transpose_f32_to_bf16<<<dim3(DM / 32, DM / 32), 256, 0, stream>>>(w_k, wk_t, DM, DM);
    transpose_f32_to_bf16<<<dim3(DM / 32, DM / 32), 256, 0, stream>>>(w_v, wv_t, DM, DM);
    transpose_f32_to_bf16<<<dim3(DM / 32, DM / 32), 256, 0, stream>>>(w_0, w0_t, DM, DM);
    transpose_f32_to_bf16<<<dim3(DFF / 32, DM / 32), 256, 0, stream>>>(w_1, w1_t, DM, DFF);
    transpose_f32_to_bf16<<<dim3(DM / 32, DFF / 32), 256, 0, stream>>>(w_2, w2_t, DFF, DM);

    // 2) LN1 -> bf16
    layernorm_bf16<<<M, 256, 0, stream>>>(x, g_1, be_1, xn1);

    // 3) Q (pre-scaled by log2(e)/sqrt(64) for base-2 softmax), K; V transposed
    dim3 g_dm(M / 128, DM / 128);
    const float qscale = 0.125f * 1.44269504088896340736f;
    gemm_wmma<OUT_BF16><<<g_dm, 256, 0, stream>>>(xn1, DM, wq_t, DM, DM, nullptr, nullptr, Qb, DM, qscale);
    gemm_wmma<OUT_BF16><<<g_dm, 256, 0, stream>>>(xn1, DM, wk_t, DM, DM, nullptr, nullptr, Kb, DM, 1.0f);
    gemm_wmma<OUT_BF16_T><<<g_dm, 256, 0, stream>>>(xn1, DM, wv_t, DM, DM, nullptr, nullptr, Vt, M, 1.0f);

    // 4) attention
    dim3 g_att(2048 / 128, 2 * 16);
    attention_wmma<<<g_att, 128, 0, stream>>>(Qb, Kb, Vt, attn);

    // 5) O projection + residual: x1 = x + attn @ w0 + b0  (f32)
    gemm_wmma<OUT_F32_RES><<<g_dm, 256, 0, stream>>>(attn, DM, w0_t, DM, DM, b_0, x, x1, DM, 1.0f);

    // 6) LN2 -> bf16
    layernorm_bf16<<<M, 256, 0, stream>>>(x1, g_2, be_2, xn2);

    // 7) FFN1: relu(xn2 @ w1 + b1) -> bf16 [M, DFF]
    dim3 g_ff1(M / 128, DFF / 128);
    gemm_wmma<OUT_BF16_RELU><<<g_ff1, 256, 0, stream>>>(xn2, DM, w1_t, DM, DM, b_1, nullptr, hmid, DFF, 1.0f);

    // 8) FFN2 + residual: out = x1 + hmid @ w2 + b2  (f32)
    gemm_wmma<OUT_F32_RES><<<g_dm, 256, 0, stream>>>(hmid, DFF, w2_t, DFF, DFF, b_2, x1, out, DM, 1.0f);

    (void)in_sizes; (void)n_in; (void)out_size; (void)ws_size;
}
